// Multihead_Self_Attention_70557722738906
// MI455X (gfx1250) — matrix operand order repaired and checked by reading, not compiled, not run on hardware
//
#include <hip/hip_runtime.h>

// ---------------------------------------------------------------------------
// Problem constants (B=4, S=2048, D=1024, H=16, DK=64)
// ---------------------------------------------------------------------------
static constexpr int kB  = 4;
static constexpr int kS  = 2048;
static constexpr int kD  = 1024;
static constexpr int kH  = 16;
static constexpr int kDK = 64;
static constexpr int kM  = kB * kS;          // 8192 rows in the projection GEMMs

typedef __attribute__((ext_vector_type(16))) __bf16 v16bf;
typedef __attribute__((ext_vector_type(8)))  float  v8f;

union FragBF {
  v16bf v;
  uint4 q[2];
};

// A-matrix 16x32 bf16 fragment (ISA 7.12.2): lane = row M, lanes 0-15 hold
// K chunks {0..7, 16..23}, lanes 16-31 hold {8..15, 24..31}.
__device__ __forceinline__ void load_a_frag(FragBF& f, const __bf16* A, int lda,
                                            int row0, int kbase, int lane) {
  const int lr = lane & 15, half = lane >> 4;
  const __bf16* p = A + (size_t)(row0 + lr) * lda + kbase + half * 8;
  f.q[0] = *(const uint4*)(p);
  f.q[1] = *(const uint4*)(p + 16);
}

// B-matrix 32x16 bf16 fragment: lane = column N, lanes 0-15 hold K chunks
// {0..7, 16..23}, lanes 16-31 hold {8..15, 24..31}.  "Bm" is stored as rows of length
// ldb indexed by N (i.e. W is (N,K) row-major), so this is a contiguous load.
__device__ __forceinline__ void load_b_frag(FragBF& f, const __bf16* Bm, int ldb,
                                            int n0, int kbase, int lane) {
  const int lr = lane & 15, half = lane >> 4;
  const __bf16* p = Bm + (size_t)(n0 + lr) * ldb + kbase + half * 8;
  f.q[0] = *(const uint4*)(p);
  f.q[1] = *(const uint4*)(p + 16);
}

__device__ __forceinline__ v8f wmma_bf16(const FragBF& a, const FragBF& b, v8f c) {
  return __builtin_amdgcn_wmma_f32_16x16x32_bf16(
      /*neg_a=*/false, a.v, /*neg_b=*/false, b.v,
      /*c_mod=*/(short)0, c, /*reuse_a=*/false, /*reuse_b=*/false);
}

// CDNA5 async global -> LDS copy (tracked by ASYNCcnt).  lds_off is the LDS
// byte offset (low 32 bits of a generic shared pointer), gaddr a 64-bit
// global address held in a VGPR pair.
__device__ __forceinline__ void async_copy_b128(uint32_t lds_off, const void* gaddr) {
  asm volatile("global_load_async_to_lds_b128 %0, %1, off"
               :: "v"(lds_off), "v"(gaddr) : "memory");
}
__device__ __forceinline__ void wait_async0() {
  asm volatile("s_wait_asynccnt 0" ::: "memory");
}

// ---------------------------------------------------------------------------
// Kernel 0: fp32 -> bf16 conversion
// ---------------------------------------------------------------------------
__global__ void __launch_bounds__(256) cvt_f32_to_bf16(const float* __restrict__ src,
                                                       __bf16* __restrict__ dst, int n) {
  int i = blockIdx.x * 256 + threadIdx.x;
  if (i < n) dst[i] = (__bf16)src[i];
}

// ---------------------------------------------------------------------------
// Kernel 1: fused QKV projection GEMM + RoPE epilogue.
//   blockIdx.z selects {Q, K, V}.  Block = 8 waves -> 128(M) x 128(N) tile,
//   each wave owns 32x64 (2x4 grid of 16x16 WMMA accumulators).
//   Q/K: RoPE applied (pair = adjacent lanes), stored bf16 [b,h,s,dk].
//   V  : stored TRANSPOSED bf16 [b,h,dk,s] so P@V B-fragments are contiguous.
// ---------------------------------------------------------------------------
__global__ void __launch_bounds__(256) qkv_gemm_rope(
    const __bf16* __restrict__ xb,
    const __bf16* __restrict__ wq, const __bf16* __restrict__ wk,
    const __bf16* __restrict__ wv,
    __bf16* __restrict__ Qb, __bf16* __restrict__ Kb, __bf16* __restrict__ Vt) {
  const int lane = threadIdx.x & 31;
  const int wave = threadIdx.x >> 5;
  const int wm = wave & 3, wn = wave >> 2;
  const int which = blockIdx.z;
  const __bf16* W = (which == 0) ? wq : (which == 1) ? wk : wv;

  const int m0 = blockIdx.x * 128 + wm * 32;
  const int n0 = blockIdx.y * 128 + wn * 64;

  v8f acc[2][4] = {};

  for (int k = 0; k < kD; k += 32) {
    FragBF a[2], b[4];
    load_a_frag(a[0], xb, kD, m0,      k, lane);
    load_a_frag(a[1], xb, kD, m0 + 16, k, lane);
#pragma unroll
    for (int nt = 0; nt < 4; ++nt)
      load_b_frag(b[nt], W, kD, n0 + nt * 16, k, lane);
#pragma unroll
    for (int mt = 0; mt < 2; ++mt)
#pragma unroll
      for (int nt = 0; nt < 4; ++nt)
        acc[mt][nt] = wmma_bf16(a[mt], b[nt], acc[mt][nt]);
  }

  // Epilogue: C/D layout -> lane holds column n, VGPR r holds row r (+8 for
  // upper half-wave).  RoPE mixes columns 2i/2i+1 == adjacent lanes.
  const int lr = lane & 15, half = lane >> 4;
#pragma unroll
  for (int nt = 0; nt < 4; ++nt) {
    const int n  = n0 + nt * 16 + lr;
    const int h  = n >> 6;        // head
    const int dk = n & 63;        // index within head
    const float inv_freq = __powf(10000.0f, -(float)(dk & 62) * (1.0f / 64.0f));
#pragma unroll
    for (int mt = 0; mt < 2; ++mt) {
#pragma unroll
      for (int r = 0; r < 8; ++r) {
        const int m  = m0 + mt * 16 + r + half * 8;
        const int s  = m & (kS - 1);
        const int bb = m >> 11;              // m / S
        float v = acc[mt][nt][r];
        if (which < 2) {
          const float p = __shfl_xor(v, 1, 32);     // RoPE partner column
          const float ang = (float)s * inv_freq;
          float sn, cs;
          __sincosf(ang, &sn, &cs);
          v = (dk & 1) ? (p * sn + v * cs) : (v * cs - p * sn);
          __bf16* dst = (which == 0) ? Qb : Kb;
          dst[(((size_t)(bb * kH + h) * kS + s) * kDK) + dk] = (__bf16)v;
        } else {
          Vt[(((size_t)(bb * kH + h) * kDK + dk) * kS) + s] = (__bf16)v;
        }
      }
    }
  }
}

// ---------------------------------------------------------------------------
// Kernel 2: causal flash attention with async double-buffered K/V staging.
//   Block = 4 waves covering a 64-row query block; all four q-tiles share the
//   same diagonal key block, so the k-loop is uniform across the block.
//   K/V 64x64 bf16 tiles are staged via global_load_async_to_lds_b128
//   (double buffered, ASYNCcnt) and consumed as WMMA fragments from LDS.
// ---------------------------------------------------------------------------
__global__ void __launch_bounds__(128) flash_attn_fwd(
    const __bf16* __restrict__ Qb, const __bf16* __restrict__ Kb,
    const __bf16* __restrict__ Vt, __bf16* __restrict__ Ob) {
  __shared__ __bf16 Kt[2][64 * 64];    // 2 x 8KB  K tiles   [j][dk]
  __shared__ __bf16 Vl[2][64 * 64];    // 2 x 8KB  V tiles   [d][j]
  __shared__ __bf16 Pl[4][16 * 64];    // 4 x 2KB  per-wave P staging

  const int tid  = threadIdx.x;
  const int lane = tid & 31;
  const int wave = tid >> 5;
  const int lr = lane & 15, half = lane >> 4;
  const int qblk = blockIdx.x;               // 0..S/64-1
  const int bh   = blockIdx.y;               // 0..B*H-1
  const int q0b  = qblk * 64;                // block's first query row
  const int q0   = q0b + wave * 16;          // this wave's q-tile

  const __bf16* Qp = Qb + (size_t)bh * kS * kDK;
  const __bf16* Kp = Kb + (size_t)bh * kS * kDK;
  const __bf16* Vp = Vt + (size_t)bh * kDK * kS;

  // Cooperative async staging of one 64-key block: 512 b128 chunks for K and
  // 128B-rows for V, 8 async copies per thread.
  auto stage = [&](int buf, int kb) {
#pragma unroll
    for (int i = 0; i < 4; ++i) {
      const int c   = tid + i * 128;         // chunk 0..511
      const int row = c >> 3;                // 0..63
      const int col = (c & 7) * 8;           // bf16 offset within row
      async_copy_b128((uint32_t)(uintptr_t)(&Kt[buf][row * 64 + col]),
                      Kp + (size_t)(kb + row) * kDK + col);
      async_copy_b128((uint32_t)(uintptr_t)(&Vl[buf][row * 64 + col]),
                      Vp + (size_t)row * kS + kb + col);
    }
  };

  FragBF aq[2];
  load_a_frag(aq[0], Qp, kDK, q0, 0,  lane);
  load_a_frag(aq[1], Qp, kDK, q0, 32, lane);

  v8f o[4] = {};
  float mrow[8], lrow[8];
#pragma unroll
  for (int r = 0; r < 8; ++r) { mrow[r] = -__builtin_inff(); lrow[r] = 0.0f; }

  const int kb_last = q0b;                   // diagonal block, same for all waves
  stage(0, 0);

  for (int kb = 0; kb <= kb_last; kb += 64) {
    const int buf = (kb >> 6) & 1;
    wait_async0();                           // own async copies landed in LDS
    __syncthreads();                         // everyone's copies landed
    if (kb + 64 <= kb_last) stage(buf ^ 1, kb + 64);   // prefetch next block

    const __bf16* Kt_ = Kt[buf];
    const __bf16* Vl_ = Vl[buf];

    // ---- scores: Q (16x64) @ K^T, 4 tiles of 16x16, K from LDS ----
    v8f sc[4];
#pragma unroll
    for (int jt = 0; jt < 4; ++jt) {
      FragBF bk;
      const __bf16* p0 = Kt_ + (jt * 16 + lr) * 64 + half * 8;
      bk.q[0] = *(const uint4*)(p0);
      bk.q[1] = *(const uint4*)(p0 + 16);
      v8f z = {};
      z = wmma_bf16(aq[0], bk, z);
      const __bf16* p1 = p0 + 32;
      bk.q[0] = *(const uint4*)(p1);
      bk.q[1] = *(const uint4*)(p1 + 16);
      z = wmma_bf16(aq[1], bk, z);
      sc[jt] = z;
    }

    const bool diag = (kb == kb_last);

    // ---- online softmax (per row = per accumulator register, half-wave) ----
#pragma unroll
    for (int r = 0; r < 8; ++r) {
      const int mq = q0 + r + half * 8;      // absolute query row
      float vals[4];
      float rmax = -__builtin_inff();
#pragma unroll
      for (int jt = 0; jt < 4; ++jt) {
        float v = sc[jt][r] * 0.125f;        // 1/sqrt(DK)
        const int j = kb + jt * 16 + lr;
        if (diag && j > mq) v = -__builtin_inff();
        vals[jt] = v;
        rmax = fmaxf(rmax, v);
      }
#pragma unroll
      for (int d = 1; d < 16; d <<= 1) rmax = fmaxf(rmax, __shfl_xor(rmax, d, 32));
      const float mnew  = fmaxf(mrow[r], rmax);
      const float alpha = __expf(mrow[r] - mnew);
      float rsum = 0.0f;
#pragma unroll
      for (int jt = 0; jt < 4; ++jt) {
        const float p = __expf(vals[jt] - mnew);   // exp(-inf) == 0
        rsum += p;
        Pl[wave][(r + half * 8) * 64 + jt * 16 + lr] = (__bf16)p;
      }
#pragma unroll
      for (int d = 1; d < 16; d <<= 1) rsum += __shfl_xor(rsum, d, 32);
      lrow[r] = lrow[r] * alpha + rsum;
      mrow[r] = mnew;
#pragma unroll
      for (int dt = 0; dt < 4; ++dt) o[dt][r] *= alpha;
    }

    // ---- O += P (16x64) @ V, A from per-wave LDS P, B from staged V tile ----
#pragma unroll
    for (int kk = 0; kk < 2; ++kk) {
      FragBF ap;
      {
        const __bf16* p = Pl[wave] + lr * 64 + kk * 32 + half * 8;
        ap.q[0] = *(const uint4*)(p);
        ap.q[1] = *(const uint4*)(p + 16);
      }
#pragma unroll
      for (int dt = 0; dt < 4; ++dt) {
        FragBF bv;
        const __bf16* p = Vl_ + (dt * 16 + lr) * 64 + kk * 32 + half * 8;
        bv.q[0] = *(const uint4*)(p);
        bv.q[1] = *(const uint4*)(p + 16);
        o[dt] = wmma_bf16(ap, bv, o[dt]);
      }
    }

    __syncthreads();    // buffer `buf` free for reuse two iterations later
  }

  // ---- normalize + store bf16 into [b, s, h*64+d] for the final GEMM ----
  const int b = bh >> 4, h = bh & 15;
#pragma unroll
  for (int r = 0; r < 8; ++r) {
    const int s = q0 + r + half * 8;
    const float inv = 1.0f / lrow[r];
#pragma unroll
    for (int dt = 0; dt < 4; ++dt) {
      const int d = dt * 16 + lr;
      Ob[((size_t)b * kS + s) * kD + h * kDK + d] = (__bf16)(o[dt][r] * inv);
    }
  }
}

// ---------------------------------------------------------------------------
// Kernel 3: output projection  out = Ob @ Wo^T  (fp32 result to d_out)
// ---------------------------------------------------------------------------
__global__ void __launch_bounds__(256) out_proj_gemm(
    const __bf16* __restrict__ Ob, const __bf16* __restrict__ wo,
    float* __restrict__ out) {
  const int lane = threadIdx.x & 31;
  const int wave = threadIdx.x >> 5;
  const int wm = wave & 3, wn = wave >> 2;
  const int m0 = blockIdx.x * 128 + wm * 32;
  const int n0 = blockIdx.y * 128 + wn * 64;

  v8f acc[2][4] = {};
  for (int k = 0; k < kD; k += 32) {
    FragBF a[2], b[4];
    load_a_frag(a[0], Ob, kD, m0,      k, lane);
    load_a_frag(a[1], Ob, kD, m0 + 16, k, lane);
#pragma unroll
    for (int nt = 0; nt < 4; ++nt)
      load_b_frag(b[nt], wo, kD, n0 + nt * 16, k, lane);
#pragma unroll
    for (int mt = 0; mt < 2; ++mt)
#pragma unroll
      for (int nt = 0; nt < 4; ++nt)
        acc[mt][nt] = wmma_bf16(a[mt], b[nt], acc[mt][nt]);
  }

  const int lr = lane & 15, half = lane >> 4;
#pragma unroll
  for (int nt = 0; nt < 4; ++nt) {
    const int n = n0 + nt * 16 + lr;
#pragma unroll
    for (int mt = 0; mt < 2; ++mt)
#pragma unroll
      for (int r = 0; r < 8; ++r) {
        const int m = m0 + mt * 16 + r + half * 8;
        out[(size_t)m * kD + n] = acc[mt][nt][r];
      }
  }
}

// ---------------------------------------------------------------------------
// Host launcher
// ---------------------------------------------------------------------------
extern "C" void kernel_launch(void* const* d_in, const int* in_sizes, int n_in,
                              void* d_out, int out_size, void* d_ws, size_t ws_size,
                              hipStream_t stream) {
  (void)in_sizes; (void)n_in; (void)out_size; (void)ws_size;
  const float* x  = (const float*)d_in[0];
  // d_in[1] = token_positions (arange(S)) -- position == s, used implicitly.
  const float* Wq = (const float*)d_in[2];
  const float* Wk = (const float*)d_in[3];
  const float* Wv = (const float*)d_in[4];
  const float* Wo = (const float*)d_in[5];
  float* out = (float*)d_out;

  // Workspace layout (bf16 elements)
  __bf16* ws = (__bf16*)d_ws;
  const size_t nX = (size_t)kM * kD;      // 8,388,608
  const size_t nW = (size_t)kD * kD;      // 1,048,576
  __bf16* xb  = ws;
  __bf16* wqb = xb  + nX;
  __bf16* wkb = wqb + nW;
  __bf16* wvb = wkb + nW;
  __bf16* wob = wvb + nW;
  __bf16* Qb  = wob + nW;                 // [B,H,S,DK]
  __bf16* Kb  = Qb  + nX;                 // [B,H,S,DK]
  __bf16* Vt  = Kb  + nX;                 // [B,H,DK,S] (transposed)
  __bf16* Ob  = Vt  + nX;                 // [B,S,D]

  // 0) fp32 -> bf16 conversions
  cvt_f32_to_bf16<<<(int)((nX + 255) / 256), 256, 0, stream>>>(x,  xb,  (int)nX);
  cvt_f32_to_bf16<<<(int)((nW + 255) / 256), 256, 0, stream>>>(Wq, wqb, (int)nW);
  cvt_f32_to_bf16<<<(int)((nW + 255) / 256), 256, 0, stream>>>(Wk, wkb, (int)nW);
  cvt_f32_to_bf16<<<(int)((nW + 255) / 256), 256, 0, stream>>>(Wv, wvb, (int)nW);
  cvt_f32_to_bf16<<<(int)((nW + 255) / 256), 256, 0, stream>>>(Wo, wob, (int)nW);

  // 1) fused QKV projection + RoPE (z = {Q,K,V})
  qkv_gemm_rope<<<dim3(kM / 128, kD / 128, 3), 256, 0, stream>>>(
      xb, wqb, wkb, wvb, Qb, Kb, Vt);

  // 2) causal flash attention (async double-buffered K/V staging)
  flash_attn_fwd<<<dim3(kS / 64, kB * kH), 128, 0, stream>>>(Qb, Kb, Vt, Ob);

  // 3) output projection
  out_proj_gemm<<<dim3(kM / 128, kD / 128), 256, 0, stream>>>(Ob, wob, out);
}
